// ComplexTransformer_20126216749268
// MI455X (gfx1250) — matrix operand order repaired and checked by reading, not compiled, not run on hardware
//
#include <hip/hip_runtime.h>

// ---------------------------------------------------------------------------
// Types
// ---------------------------------------------------------------------------
typedef __bf16 bf16_t;
typedef __attribute__((ext_vector_type(16))) __bf16    bf16x16;
typedef __attribute__((ext_vector_type(8)))  float     f32x8;
typedef __attribute__((ext_vector_type(4)))  unsigned  u32x4;
typedef __attribute__((ext_vector_type(8)))  int       i32x8;
typedef __attribute__((ext_vector_type(4)))  int       i32x4;

// Problem constants (from the reference)
#define TOK   2048      // B*N
#define DIMC  512
#define FFC   2048
#define NSEQ  1024
#define NBH   16        // B*HEADS
#define DHEAD 64

// ---------------------------------------------------------------------------
// Tensor Data Mover (gfx1250). Arity differs between toolchains:
// ROCm 7.2 / clang-22: 5 args; amdgpu-toolchain / clang-23: 6 args.
// ---------------------------------------------------------------------------
#if defined(__has_builtin)
#if __has_builtin(__builtin_amdgcn_tensor_load_to_lds)
#define HAVE_TDM 1
#endif
#endif

#ifdef HAVE_TDM
// 2D tile load: tile_h rows of tile_w bf16 elements, row stride = stride
// elements, from a tensor of tensor_w x tensor_h bf16, into LDS at lds_off.
static __device__ __forceinline__ void tdm_load_2d(unsigned lds_off,
                                                   const void* gptr,
                                                   unsigned tile_w, unsigned tile_h,
                                                   unsigned tensor_w, unsigned tensor_h,
                                                   unsigned stride) {
  unsigned long long ga = (unsigned long long)gptr;
  u32x4 g0;
  g0[0] = 1u;                                          // count=1, user D#
  g0[1] = lds_off;                                     // lds_addr (bytes)
  g0[2] = (unsigned)(ga & 0xFFFFFFFFu);                // global_addr lo
  g0[3] = (unsigned)((ga >> 32) & 0x1FFFFFFu) | (2u << 30);  // addr hi | type=2
  i32x8 g1;
  g1[0] = (int)(1u << 16);                             // data_size=1 (2 bytes)
  g1[1] = (int)((tensor_w & 0xFFFFu) << 16);           // tensor_dim0[15:0]
  g1[2] = (int)((tensor_w >> 16) | ((tensor_h & 0xFFFFu) << 16));
  g1[3] = (int)((tensor_h >> 16) | (tile_w << 16));    // | tile_dim0
  g1[4] = (int)(tile_h & 0xFFFFu);                     // tile_dim1 (tile_dim2=0)
  g1[5] = (int)stride;                                 // tensor_dim0_stride lo
  g1[6] = 0;
  g1[7] = 0;
  i32x4 gz = {0, 0, 0, 0};
#if __clang_major__ >= 23
  i32x8 gz8 = {0, 0, 0, 0, 0, 0, 0, 0};
  __builtin_amdgcn_tensor_load_to_lds(g0, g1, gz, gz, gz8, 0);
#else
  __builtin_amdgcn_tensor_load_to_lds(g0, g1, gz, gz, 0);
#endif
}
static __device__ __forceinline__ unsigned lds_off_of(const void* p) {
  return (unsigned)(unsigned long long)p;   // LDS aperture: addr[31:0]
}
#endif

// ---------------------------------------------------------------------------
// WMMA helpers (wave32, v_wmma_f32_16x16x32_bf16)
// ---------------------------------------------------------------------------
static __device__ __forceinline__ f32x8 wmma_bf(bf16x16 a, bf16x16 b, f32x8 c) {
  return __builtin_amdgcn_wmma_f32_16x16x32_bf16(false, a, false, b, (short)0, c,
                                                 false, false);
}

// A fragment: 16x32 bf16 row-major, stride in elements.
static __device__ __forceinline__ bf16x16 load_frag_a(const bf16_t* p, int stride) {
  int lane = threadIdx.x & 31;
  int row  = lane & 15;
  int k0   = (lane < 16) ? 0 : 16;
  const bf16_t* r = p + (size_t)row * stride + k0;
  union { bf16x16 v; u32x4 q[2]; } u;
  u.q[0] = *(const u32x4*)r;          // K k0..k0+7
  u.q[1] = *(const u32x4*)(r + 8);    // K k0+8..k0+15
  return u.v;
}

// B fragment from B^T storage (N x K row-major): lane n holds column n,
// lane<16 -> K 0..15, lane>=16 -> K 16..31 (element i == K k0+i).
static __device__ __forceinline__ bf16x16 load_frag_b(const bf16_t* p, int stride) {
  int lane = threadIdx.x & 31;
  int n    = lane & 15;
  int k0   = (lane < 16) ? 0 : 16;
  const bf16_t* r = p + (size_t)n * stride + k0;
  union { bf16x16 v; u32x4 q[2]; } u;
  u.q[0] = *(const u32x4*)r;
  u.q[1] = *(const u32x4*)(r + 8);
  return u.v;
}

// B fragment from K x N row-major storage (strided per-element gather).
static __device__ __forceinline__ bf16x16 load_frag_b_kxn(const bf16_t* p, int stride) {
  int lane = threadIdx.x & 31;
  int n    = lane & 15;
  int k0   = (lane < 16) ? 0 : 16;
  bf16x16 v;
#pragma unroll
  for (int i = 0; i < 16; ++i) v[i] = p[(size_t)(k0 + i) * stride + n];
  return v;
}

// Negate a bf16 fragment (flip sign bits).
static __device__ __forceinline__ bf16x16 neg_frag(bf16x16 f) {
  union { bf16x16 v; unsigned u[8]; } x;
  x.v = f;
#pragma unroll
  for (int i = 0; i < 8; ++i) x.u[i] ^= 0x80008000u;
  return x.v;
}

// ---------------------------------------------------------------------------
// Weight repack: f32 interleaved complex (K,Nc,2) -> transposed bf16 planes.
// ---------------------------------------------------------------------------
__global__ void pack_w_kernel(const float* __restrict__ w, bf16_t* __restrict__ wtr,
                              bf16_t* __restrict__ wti, int K, int Nc) {
  int idx = blockIdx.x * 256 + threadIdx.x;
  if (idx >= K * Nc) return;
  int k = idx / Nc, n = idx - k * Nc;
  wtr[(size_t)n * K + k] = (bf16_t)w[(size_t)idx * 2 + 0];
  wti[(size_t)n * K + k] = (bf16_t)w[(size_t)idx * 2 + 1];
}

// ---------------------------------------------------------------------------
// Complex RMSNorm. mode 0: bf16 planes out; mode 1: f32 interleaved out.
// ---------------------------------------------------------------------------
__global__ __launch_bounds__(256)
void rmsnorm_kernel(const float* __restrict__ xc, const float* __restrict__ gamma,
                    bf16_t* __restrict__ hr, bf16_t* __restrict__ hi,
                    float* __restrict__ outf, int mode) {
  __shared__ float red[8];
  __shared__ float invs;
  int m = blockIdx.x;
  int t = threadIdx.x;
  float s = 0.f;
  for (int d = t; d < DIMC; d += 256) {
    float xr = xc[((size_t)m * DIMC + d) * 2 + 0];
    float xi = xc[((size_t)m * DIMC + d) * 2 + 1];
    s += xr * xr + xi * xi;
  }
  for (int msk = 16; msk; msk >>= 1) s += __shfl_xor(s, msk, 32);
  if ((t & 31) == 0) red[t >> 5] = s;
  __syncthreads();
  if (t == 0) {
    float tot = 0.f;
    for (int i = 0; i < 8; ++i) tot += red[i];
    invs = rsqrtf(tot * (1.0f / (float)DIMC) + 1e-6f);
  }
  __syncthreads();
  float inv = invs;
  for (int d = t; d < DIMC; d += 256) {
    float xr = xc[((size_t)m * DIMC + d) * 2 + 0] * inv;
    float xi = xc[((size_t)m * DIMC + d) * 2 + 1] * inv;
    float gr = gamma[d * 2 + 0], gi = gamma[d * 2 + 1];
    float or_ = xr * gr - xi * gi;
    float oi_ = xr * gi + xi * gr;
    if (mode == 0) {
      hr[(size_t)m * DIMC + d] = (bf16_t)or_;
      hi[(size_t)m * DIMC + d] = (bf16_t)oi_;
    } else {
      outf[((size_t)m * DIMC + d) * 2 + 0] = or_;
      outf[((size_t)m * DIMC + d) * 2 + 1] = oi_;
    }
  }
}

// ---------------------------------------------------------------------------
// Complex GEMM: C[M,Nc] = A[M,K]*W[K,Nc] (+bias). Register-blocked: each wave
// owns a 16x64 output strip (4 N-tiles) -> 16 WMMAs per 32-K step; the shared
// 16x32 A tile (r+i planes) is DMA'd into LDS by the Tensor Data Mover.
// Block = 256 threads (8 waves) covers 16 x 512 output.
// ---------------------------------------------------------------------------
__global__ __launch_bounds__(256)
void cgemm_kernel(const bf16_t* __restrict__ ar, const bf16_t* __restrict__ ai,
                  const bf16_t* __restrict__ btr, const bf16_t* __restrict__ bti,
                  const float* __restrict__ bias,
                  float* __restrict__ cr, float* __restrict__ ci,
                  int K, int Nc) {
  __shared__ __attribute__((aligned(16))) bf16_t lds_a[2][16][32];
  const int wave = threadIdx.x >> 5;
  const int lane = threadIdx.x & 31;
  const int m0 = blockIdx.x * 16;
  const int n0 = blockIdx.y * 512 + wave * 64;

  f32x8 zero = {0, 0, 0, 0, 0, 0, 0, 0};
  f32x8 accr[4], acci[4];
#pragma unroll
  for (int t = 0; t < 4; ++t) { accr[t] = zero; acci[t] = zero; }

  for (int k0 = 0; k0 < K; k0 += 32) {
    __syncthreads();
#ifdef HAVE_TDM
    if (wave == 0) {
      tdm_load_2d(lds_off_of(&lds_a[0][0][0]), ar + (size_t)m0 * K + k0,
                  32, 16, (unsigned)K, (unsigned)TOK, (unsigned)K);
      tdm_load_2d(lds_off_of(&lds_a[1][0][0]), ai + (size_t)m0 * K + k0,
                  32, 16, (unsigned)K, (unsigned)TOK, (unsigned)K);
      __builtin_amdgcn_s_wait_tensorcnt(0);
    }
#else
    {
      int idx = threadIdx.x * 2;          // 512 bf16 per plane, 2 per thread
      int row = idx >> 5, col = idx & 31;
      *(unsigned*)&lds_a[0][row][col] =
          *(const unsigned*)(ar + (size_t)(m0 + row) * K + k0 + col);
      *(unsigned*)&lds_a[1][row][col] =
          *(const unsigned*)(ai + (size_t)(m0 + row) * K + k0 + col);
    }
#endif
    __syncthreads();

    bf16x16 A_r  = load_frag_a(&lds_a[0][0][0], 32);
    bf16x16 A_i  = load_frag_a(&lds_a[1][0][0], 32);
    bf16x16 A_in = neg_frag(A_i);

#pragma unroll
    for (int t = 0; t < 4; ++t) {
      const bf16_t* brp = btr + (size_t)(n0 + t * 16) * K + k0;
      const bf16_t* bip = bti + (size_t)(n0 + t * 16) * K + k0;
      if (t == 0 && k0 + 32 < K) {
        __builtin_prefetch(brp + 32, 0, 1);
        __builtin_prefetch(bip + 32, 0, 1);
      }
      bf16x16 B_r = load_frag_b(brp, K);
      bf16x16 B_i = load_frag_b(bip, K);
      accr[t] = wmma_bf(A_r,  B_r, accr[t]);
      accr[t] = wmma_bf(A_in, B_i, accr[t]);   // re -= ai*bi
      acci[t] = wmma_bf(A_r,  B_i, acci[t]);
      acci[t] = wmma_bf(A_i,  B_r, acci[t]);
    }
  }

  int n  = lane & 15;
  int mh = (lane < 16) ? 0 : 8;
#pragma unroll
  for (int t = 0; t < 4; ++t) {
    int nc = n0 + t * 16 + n;
    float biasr = 0.f, biasi = 0.f;
    if (bias) { biasr = bias[nc * 2 + 0]; biasi = bias[nc * 2 + 1]; }
#pragma unroll
    for (int i = 0; i < 8; ++i) {
      size_t o = (size_t)(m0 + mh + i) * Nc + nc;
      cr[o] = accr[t][i] + biasr;
      ci[o] = acci[t][i] + biasi;
    }
  }
}

// ---------------------------------------------------------------------------
// RoPE + head split + bf16 repack.
// ---------------------------------------------------------------------------
__global__ void rope_kernel(const float* __restrict__ qr_, const float* __restrict__ qi_,
                            const float* __restrict__ kvr_, const float* __restrict__ kvi_,
                            bf16_t* __restrict__ qhr, bf16_t* __restrict__ qhi,
                            bf16_t* __restrict__ khr, bf16_t* __restrict__ khi,
                            bf16_t* __restrict__ vhr, bf16_t* __restrict__ vhi) {
  int idx = blockIdx.x * 256 + threadIdx.x;
  if (idx >= TOK * DIMC) return;
  int e = idx & (DIMC - 1);
  int m = idx >> 9;
  int b = m >> 10;
  int n = m & (NSEQ - 1);
  int h = e >> 6;
  int d = e & (DHEAD - 1);
  float invf = __powf(10000.0f, -(float)d * (1.0f / (float)DHEAD));
  float sn, cs;
  __sincosf((float)n * invf, &sn, &cs);
  size_t src = (size_t)m * DIMC + e;
  size_t dst = (((size_t)(b * 8 + h)) * NSEQ + n) * DHEAD + d;
  float qr = qr_[src], qi = qi_[src];
  qhr[dst] = (bf16_t)(qr * cs - qi * sn);
  qhi[dst] = (bf16_t)(qr * sn + qi * cs);
  float kr = kvr_[(size_t)m * (2 * DIMC) + e];
  float ki = kvi_[(size_t)m * (2 * DIMC) + e];
  khr[dst] = (bf16_t)(kr * cs - ki * sn);
  khi[dst] = (bf16_t)(kr * sn + ki * cs);
  vhr[dst] = (bf16_t)kvr_[(size_t)m * (2 * DIMC) + DIMC + e];
  vhi[dst] = (bf16_t)kvi_[(size_t)m * (2 * DIMC) + DIMC + e];
}

// ---------------------------------------------------------------------------
// Flash attention for one of the 4 real combos (rr/ri/ir/ii).
// grid = (N/128, B*H, 4); block = 256 (8 waves); each wave: 16 query rows.
// K/V 32-key chunks are DMA'd into LDS (TDM, pure contiguous copy); the
// -ki sign flip is applied on the loaded B fragment instead.
// ---------------------------------------------------------------------------
__global__ __launch_bounds__(256)
void attn_kernel(const bf16_t* __restrict__ qhr, const bf16_t* __restrict__ qhi,
                 const bf16_t* __restrict__ khr, const bf16_t* __restrict__ khi,
                 const bf16_t* __restrict__ vhr, const bf16_t* __restrict__ vhi,
                 float* __restrict__ ocomb) {
  __shared__ __attribute__((aligned(16))) bf16_t lds_k[32][64];
  __shared__ __attribute__((aligned(16))) bf16_t lds_v[32][64];
  __shared__ __attribute__((aligned(16))) bf16_t lds_p[8][16][32];

  const int r    = blockIdx.z;
  const int bh   = blockIdx.y;
  const int wave = threadIdx.x >> 5;
  const int lane = threadIdx.x & 31;
  const int q0   = blockIdx.x * 128 + wave * 16;

  // qs=[qr,qr,qi,qi]  ks=[kr,-ki,kr,-ki]  vs=[vr,vr,vi,vi]
  const bf16_t* qp = (r < 2) ? qhr : qhi;
  const bf16_t* kp = (r & 1) ? khi : khr;
  const bf16_t* vp = (r < 2) ? vhr : vhi;
  const bool knegate = (r & 1) != 0;

  const bf16_t* qbase = qp + ((size_t)bh * NSEQ + q0) * DHEAD;
  bf16x16 aq0 = load_frag_a(qbase, DHEAD);        // d 0..31
  bf16x16 aq1 = load_frag_a(qbase + 32, DHEAD);   // d 32..63

  float rm[8], rs[8];
  f32x8 oa[4];
  f32x8 zero = {0, 0, 0, 0, 0, 0, 0, 0};
#pragma unroll
  for (int i = 0; i < 8; ++i) { rm[i] = -3.0e38f; rs[i] = 0.f; }
#pragma unroll
  for (int c = 0; c < 4; ++c) oa[c] = zero;

  for (int j0 = 0; j0 < NSEQ; j0 += 32) {
    size_t gbase = ((size_t)bh * NSEQ + j0) * DHEAD;   // chunk is contiguous
    __syncthreads();
#ifdef HAVE_TDM
    if (wave == 0) {
      tdm_load_2d(lds_off_of(&lds_k[0][0]), kp + gbase, 2048, 1,
                  (unsigned)(NBH * NSEQ * DHEAD), 1, 2048);
      tdm_load_2d(lds_off_of(&lds_v[0][0]), vp + gbase, 2048, 1,
                  (unsigned)(NBH * NSEQ * DHEAD), 1, 2048);
      __builtin_amdgcn_s_wait_tensorcnt(0);
    }
#else
    {
      int t = threadIdx.x;
      int key = t >> 3;
      int d   = (t & 7) * 8;
      *(u32x4*)&lds_k[key][d] = *(const u32x4*)(kp + gbase + key * DHEAD + d);
      *(u32x4*)&lds_v[key][d] = *(const u32x4*)(vp + gbase + key * DHEAD + d);
    }
#endif
    __syncthreads();

    // S = Q K^T  (two 16x16 key sub-tiles, K-dim 64 = 2 wmma steps each)
    f32x8 s0 = zero, s1 = zero;
    bf16x16 bk;
    bk = load_frag_b(&lds_k[0][0], 64);   if (knegate) bk = neg_frag(bk);
    s0 = wmma_bf(aq0, bk, s0);
    bk = load_frag_b(&lds_k[0][32], 64);  if (knegate) bk = neg_frag(bk);
    s0 = wmma_bf(aq1, bk, s0);
    bk = load_frag_b(&lds_k[16][0], 64);  if (knegate) bk = neg_frag(bk);
    s1 = wmma_bf(aq0, bk, s1);
    bk = load_frag_b(&lds_k[16][32], 64); if (knegate) bk = neg_frag(bk);
    s1 = wmma_bf(aq1, bk, s1);

    int n  = lane & 15;
    int mh = (lane < 16) ? 0 : 8;
#pragma unroll
    for (int i = 0; i < 8; ++i) {
      float v0 = s0[i] * 0.125f;   // SCALE = DH^-0.5 = 1/8
      float v1 = s1[i] * 0.125f;
      float cmx = fmaxf(v0, v1);
      cmx = fmaxf(cmx, __shfl_xor(cmx, 1, 32));
      cmx = fmaxf(cmx, __shfl_xor(cmx, 2, 32));
      cmx = fmaxf(cmx, __shfl_xor(cmx, 4, 32));
      cmx = fmaxf(cmx, __shfl_xor(cmx, 8, 32));
      float mnew = fmaxf(rm[i], cmx);
      float corr = __expf(rm[i] - mnew);
      float p0 = __expf(v0 - mnew);
      float p1 = __expf(v1 - mnew);
      float ps = p0 + p1;
      ps += __shfl_xor(ps, 1, 32);
      ps += __shfl_xor(ps, 2, 32);
      ps += __shfl_xor(ps, 4, 32);
      ps += __shfl_xor(ps, 8, 32);
      rs[i] = rs[i] * corr + ps;
      rm[i] = mnew;
#pragma unroll
      for (int c = 0; c < 4; ++c) oa[c][i] *= corr;
      int row = mh + i;
      lds_p[wave][row][n]      = (bf16_t)p0;
      lds_p[wave][row][16 + n] = (bf16_t)p1;
    }

    // O += P V  (per-wave LDS round-trip re-layouts P as an A fragment;
    // same-wave DS ops are in-order, no block barrier needed)
    bf16x16 ap = load_frag_a(&lds_p[wave][0][0], 32);
#pragma unroll
    for (int c = 0; c < 4; ++c) {
      bf16x16 bv = load_frag_b_kxn(&lds_v[0][c * 16], 64);
      oa[c] = wmma_bf(ap, bv, oa[c]);
    }
  }

  int n  = lane & 15;
  int mh = (lane < 16) ? 0 : 8;
  float* oc = ocomb + (size_t)r * ((size_t)NBH * NSEQ * DHEAD);
#pragma unroll
  for (int i = 0; i < 8; ++i) {
    float inv = 1.0f / rs[i];
#pragma unroll
    for (int c = 0; c < 4; ++c) {
      oc[((size_t)bh * NSEQ + q0 + mh + i) * DHEAD + c * 16 + n] = oa[c][i] * inv;
    }
  }
}

// ---------------------------------------------------------------------------
// Combine the 4 attention combos: real = o0 - o3, imag = o1 + o2.
// ---------------------------------------------------------------------------
__global__ void combine_kernel(const float* __restrict__ oc,
                               bf16_t* __restrict__ or_, bf16_t* __restrict__ oi_) {
  int idx = blockIdx.x * 256 + threadIdx.x;
  if (idx >= TOK * DIMC) return;
  int e = idx & (DIMC - 1);
  int m = idx >> 9;
  int b = m >> 10;
  int n = m & (NSEQ - 1);
  int h = e >> 6;
  int d = e & (DHEAD - 1);
  size_t src = (((size_t)(b * 8 + h)) * NSEQ + n) * DHEAD + d;
  const size_t plane = (size_t)NBH * NSEQ * DHEAD;
  float o0 = oc[src];
  float o1 = oc[plane + src];
  float o2 = oc[2 * plane + src];
  float o3 = oc[3 * plane + src];
  or_[idx] = (bf16_t)(o0 - o3);
  oi_[idx] = (bf16_t)(o1 + o2);
}

// ---------------------------------------------------------------------------
// Residual add into f32 interleaved state.
// ---------------------------------------------------------------------------
__global__ void residual_kernel(float* __restrict__ xc,
                                const float* __restrict__ cr,
                                const float* __restrict__ ci) {
  int idx = blockIdx.x * 256 + threadIdx.x;
  if (idx >= TOK * DIMC) return;
  xc[(size_t)idx * 2 + 0] += cr[idx];
  xc[(size_t)idx * 2 + 1] += ci[idx];
}

// ---------------------------------------------------------------------------
// FFN nonlinearity: relu(|h|+bias)^2 * exp(i*angle(h)).
// ---------------------------------------------------------------------------
__global__ void nonlin_kernel(const float* __restrict__ gr_, const float* __restrict__ gi_,
                              const float* __restrict__ mod_bias, int l,
                              bf16_t* __restrict__ fr, bf16_t* __restrict__ fi) {
  int idx = blockIdx.x * 256 + threadIdx.x;
  if (idx >= TOK * FFC) return;
  float gr = gr_[idx], gi = gi_[idx];
  float mag = sqrtf(gr * gr + gi * gi);
  float cl = fmaxf(mag + mod_bias[l], 0.0f);
  cl = cl * cl;
  float fac = (mag > 0.0f) ? (cl / mag) : 0.0f;
  fr[idx] = (bf16_t)(gr * fac);
  fi[idx] = (bf16_t)(gi * fac);
}

// ---------------------------------------------------------------------------
// Orchestration
// ---------------------------------------------------------------------------
extern "C" void kernel_launch(void* const* d_in, const int* in_sizes, int n_in,
                              void* d_out, int out_size, void* d_ws, size_t ws_size,
                              hipStream_t stream) {
  const float* x          = (const float*)d_in[0];
  const float* gamma_attn = (const float*)d_in[1];
  const float* Wq         = (const float*)d_in[2];
  const float* Wkv        = (const float*)d_in[3];
  const float* Wo         = (const float*)d_in[4];
  const float* gamma_ff   = (const float*)d_in[5];
  const float* W1         = (const float*)d_in[6];
  const float* b1         = (const float*)d_in[7];
  const float* mod_bias   = (const float*)d_in[8];
  const float* W2         = (const float*)d_in[9];
  const float* b2         = (const float*)d_in[10];
  const float* gamma_fin  = (const float*)d_in[11];

  char* w = (char*)d_ws;
  size_t off = 0;
  auto carve = [&](size_t bytes) -> void* {
    void* p = w + off;
    off += (bytes + 255) & ~(size_t)255;
    return p;
  };
  float*  XC  = (float*)carve((size_t)TOK * DIMC * 2 * 4);
  bf16_t* HR  = (bf16_t*)carve((size_t)TOK * DIMC * 2);
  bf16_t* HI  = (bf16_t*)carve((size_t)TOK * DIMC * 2);
  bf16_t* WTR = (bf16_t*)carve((size_t)FFC * DIMC * 2);
  bf16_t* WTI = (bf16_t*)carve((size_t)FFC * DIMC * 2);
  float*  GR  = (float*)carve((size_t)TOK * FFC * 4);
  float*  GI  = (float*)carve((size_t)TOK * FFC * 4);
  bf16_t* QHR = (bf16_t*)carve((size_t)TOK * DIMC * 2);
  bf16_t* QHI = (bf16_t*)carve((size_t)TOK * DIMC * 2);
  bf16_t* KHR = (bf16_t*)carve((size_t)TOK * DIMC * 2);
  bf16_t* KHI = (bf16_t*)carve((size_t)TOK * DIMC * 2);
  bf16_t* VHR = (bf16_t*)carve((size_t)TOK * DIMC * 2);
  bf16_t* VHI = (bf16_t*)carve((size_t)TOK * DIMC * 2);
  float*  OC  = (float*)carve((size_t)4 * NBH * NSEQ * DHEAD * 4);
  bf16_t* FFR = (bf16_t*)OC;                                    // reuse (dead)
  bf16_t* FFI = (bf16_t*)((char*)OC + (size_t)TOK * FFC * 2);
  float* QR  = GR;
  float* QI  = GI;
  float* KVR = (float*)((char*)GR + (size_t)TOK * DIMC * 4);
  float* KVI = (float*)((char*)GI + (size_t)TOK * DIMC * 4);

  const int EW = (TOK * DIMC + 255) / 256;

  hipMemcpyAsync(XC, x, (size_t)TOK * DIMC * 2 * 4, hipMemcpyDeviceToDevice, stream);

  for (int l = 0; l < 2; ++l) {
    const float* ga = gamma_attn + (size_t)l * DIMC * 2;
    const float* gf = gamma_ff + (size_t)l * DIMC * 2;

    // --- attention block ---
    rmsnorm_kernel<<<TOK, 256, 0, stream>>>(XC, ga, HR, HI, nullptr, 0);

    pack_w_kernel<<<(DIMC * DIMC + 255) / 256, 256, 0, stream>>>(
        Wq + (size_t)l * DIMC * DIMC * 2, WTR, WTI, DIMC, DIMC);
    cgemm_kernel<<<dim3(TOK / 16, DIMC / 512), 256, 0, stream>>>(
        HR, HI, WTR, WTI, nullptr, QR, QI, DIMC, DIMC);

    pack_w_kernel<<<(DIMC * 2 * DIMC + 255) / 256, 256, 0, stream>>>(
        Wkv + (size_t)l * DIMC * 2 * DIMC * 2, WTR, WTI, DIMC, 2 * DIMC);
    cgemm_kernel<<<dim3(TOK / 16, (2 * DIMC) / 512), 256, 0, stream>>>(
        HR, HI, WTR, WTI, nullptr, KVR, KVI, DIMC, 2 * DIMC);

    rope_kernel<<<EW, 256, 0, stream>>>(QR, QI, KVR, KVI,
                                        QHR, QHI, KHR, KHI, VHR, VHI);

    attn_kernel<<<dim3(NSEQ / 128, NBH, 4), 256, 0, stream>>>(
        QHR, QHI, KHR, KHI, VHR, VHI, OC);

    combine_kernel<<<EW, 256, 0, stream>>>(OC, HR, HI);

    pack_w_kernel<<<(DIMC * DIMC + 255) / 256, 256, 0, stream>>>(
        Wo + (size_t)l * DIMC * DIMC * 2, WTR, WTI, DIMC, DIMC);
    cgemm_kernel<<<dim3(TOK / 16, DIMC / 512), 256, 0, stream>>>(
        HR, HI, WTR, WTI, nullptr, GR, GI, DIMC, DIMC);

    residual_kernel<<<EW, 256, 0, stream>>>(XC, GR, GI);

    // --- FFN block ---
    rmsnorm_kernel<<<TOK, 256, 0, stream>>>(XC, gf, HR, HI, nullptr, 0);

    pack_w_kernel<<<(DIMC * FFC + 255) / 256, 256, 0, stream>>>(
        W1 + (size_t)l * DIMC * FFC * 2, WTR, WTI, DIMC, FFC);
    cgemm_kernel<<<dim3(TOK / 16, FFC / 512), 256, 0, stream>>>(
        HR, HI, WTR, WTI, b1 + (size_t)l * FFC * 2, GR, GI, DIMC, FFC);

    nonlin_kernel<<<(TOK * FFC + 255) / 256, 256, 0, stream>>>(
        GR, GI, mod_bias, l, FFR, FFI);

    pack_w_kernel<<<(FFC * DIMC + 255) / 256, 256, 0, stream>>>(
        W2 + (size_t)l * FFC * DIMC * 2, WTR, WTI, FFC, DIMC);
    cgemm_kernel<<<dim3(TOK / 16, DIMC / 512), 256, 0, stream>>>(
        FFR, FFI, WTR, WTI, b2 + (size_t)l * DIMC * 2, GR, GI, FFC, DIMC);

    residual_kernel<<<EW, 256, 0, stream>>>(XC, GR, GI);
  }

  rmsnorm_kernel<<<TOK, 256, 0, stream>>>(XC, gamma_fin, nullptr, nullptr,
                                          (float*)d_out, 1);
}
